// GraphAttentionLayer_13958643712841
// MI455X (gfx1250) — matrix operand order repaired and checked by reading, not compiled, not run on hardware
//
#include <hip/hip_runtime.h>
#include <hip/hip_bf16.h>

#define N_ROWS 8192
#define F_IN   128
#define F_OUT  64
#define GAT_ALPHA 0.2f
#define TJ     128            // j-rows staged in LDS per block iteration
#define TJP    (TJ + 8)       // padded LDS row stride (halves) -> bank-conflict free B reads
#define NSTAGE (N_ROWS / TJ)  // 64

typedef __attribute__((ext_vector_type(16))) _Float16 v16h;
typedef __attribute__((ext_vector_type(8)))  _Float16 v8h;
typedef __attribute__((ext_vector_type(8)))  float    v8f;
typedef __attribute__((ext_vector_type(4)))  unsigned int u32x4;
typedef __attribute__((ext_vector_type(8)))  int      i32x8;
typedef __attribute__((ext_vector_type(4)))  int      i32x4;

#if __has_builtin(__builtin_amdgcn_tensor_load_to_lds)
#define USE_TDM 1
#else
#define USE_TDM 0
#endif

// ---------------- Kernel 1: h = X @ W  (f32), plus fp16 transposed copy hT[f][i]
__global__ void k_proj(const float* __restrict__ X, const float* __restrict__ W,
                       float* __restrict__ hf32, _Float16* __restrict__ hT) {
    int idx = blockIdx.x * blockDim.x + threadIdx.x;    // N_ROWS*F_OUT threads
    int i = idx >> 6;
    int f = idx & 63;
    const float* xr = X + i * F_IN;
    const float* wc = W + f;
    float acc = 0.f;
#pragma unroll 8
    for (int k = 0; k < F_IN; ++k)
        acc = fmaf(xr[k], wc[k * F_OUT], acc);
    hf32[idx] = acc;
    hT[f * N_ROWS + i] = (_Float16)acc;
}

// ---------------- Kernel 2: f1 = h@a1, f2 = h@a2
__global__ void k_f12(const float* __restrict__ hf32, const float* __restrict__ a,
                      float* __restrict__ f1, float* __restrict__ f2) {
    int i = blockIdx.x * blockDim.x + threadIdx.x;
    if (i >= N_ROWS) return;
    const float* hr = hf32 + i * F_OUT;
    float s1 = 0.f, s2 = 0.f;
#pragma unroll 8
    for (int j = 0; j < F_OUT; ++j) {
        float hv = hr[j];
        s1 = fmaf(hv, a[j], s1);
        s2 = fmaf(hv, a[F_OUT + j], s2);
    }
    f1[i] = s1;
    f2[i] = s2;
}

// ---------------- Kernel 3: f2max = max(f2)   (softmax row-max is lrelu(f1_i + f2max))
__global__ void k_max(const float* __restrict__ f2, float* __restrict__ f2max) {
    __shared__ float sm[256];
    float m = -INFINITY;
    for (int i = threadIdx.x; i < N_ROWS; i += 256) m = fmaxf(m, f2[i]);
    sm[threadIdx.x] = m;
    __syncthreads();
    for (int s = 128; s > 0; s >>= 1) {
        if (threadIdx.x < s) sm[threadIdx.x] = fmaxf(sm[threadIdx.x], sm[threadIdx.x + s]);
        __syncthreads();
    }
    if (threadIdx.x == 0) *f2max = sm[0];
}

#if USE_TDM
// Issue a TDM 2-D tile load:  hT[0..63][jt .. jt+TJ) -> LDS at lds_byte_off,
// with hardware padding of 4 DWORDs (16 B) after every 64 DWORDs (one 128-half
// row) so the LDS image has row stride TJP halves (bank-conflict-free B reads).
__device__ __forceinline__ void tdm_issue(const _Float16* hT, int jt,
                                          unsigned int lds_byte_off) {
    unsigned long long ga = (unsigned long long)(size_t)(hT + jt);
    u32x4 g0;
    g0[0] = 1u;                                   // count=1 (valid), flags 0
    g0[1] = lds_byte_off;                         // lds_addr
    g0[2] = (unsigned int)ga;                     // global_addr[31:0]
    g0[3] = (unsigned int)((ga >> 32) & 0x1FFFFFFu) | (2u << 30);  // addr[56:32] | type=2
    i32x8 g1;
    g1[0] = (1 << 16)        // data_size = 1 (2 bytes)
          | (1 << 20)        // pad_enable
          | (5 << 22)        // pad_interval = 64 DWORDs (256 B = one tile row)
          | (3 << 25);       // pad_amount   = 4 DWORDs (16 B)
    g1[1] = (int)(0x2000u << 16);  // tensor_dim0[15:0] = 8192 (elements)
    g1[2] = (64 << 16);            // tensor_dim0 hi = 0 | tensor_dim1 = 64
    g1[3] = (int)(128u << 16);     // tensor_dim1 hi = 0 | tile_dim0 = 128
    g1[4] = 64;                    // tile_dim1 = 64, tile_dim2 = 0
    g1[5] = 8192;                  // tensor_dim0_stride[31:0] = 8192 elements
    g1[6] = 0;
    g1[7] = 0;
    i32x4 gz4 = {0, 0, 0, 0};                     // groups 2/3 unused (2-D tensor)
    i32x8 gz8 = {0, 0, 0, 0, 0, 0, 0, 0};
    // amdgpu-toolchain (clang-23) 6-arg form:
    // (uint32x4 g0, int32x8 g1, int32x4, int32x4, int32x8, i32 cpol)
    __builtin_amdgcn_tensor_load_to_lds(g0, g1, gz4, gz4, gz8, 0);
}
#endif

// ---------------- Kernel 4: fused  softmax(leaky_relu(f1_i+f2_j)) @ h  -> elu
// Block: 128 threads (4 waves). Wave w handles rows [blk*64 + w*16, +16).
// Double-buffered LDS; TDM (wave 0) prefetches tile s+1 while all waves run
// 16 wmma + 64 exp on tile s.
__global__ void __launch_bounds__(128)
k_attn(const _Float16* __restrict__ hT, const float* __restrict__ f1,
       const float* __restrict__ f2, const float* __restrict__ f2maxp,
       float* __restrict__ out) {
    __shared__ __align__(16) _Float16 ldsH[2][F_OUT * TJP];
    __shared__ float ldsF[2][TJ];

    const int tid  = threadIdx.x;
    const int wave = tid >> 5;
    const int lane = tid & 31;
    const int half = lane >> 4;   // which 16-lane half of the wave
    const int nlo  = lane & 15;
    const int i0   = blockIdx.x * 64 + wave * 16;
    const int irow = i0 + nlo;    // A-matrix row owned by this lane

    const float f1i = f1[irow];
    const float f2m = *f2maxp;
    float mi = f1i + f2m;
    mi = (mi > 0.f) ? mi : GAT_ALPHA * mi;   // row max of e (lrelu monotonic)

    v8f cacc[4] = {v8f{}, v8f{}, v8f{}, v8f{}};
    float zacc = 0.f;

#if USE_TDM
    // Prologue: stage tile 0.
    ldsF[0][tid] = f2[tid];
    if (wave == 0)
        tdm_issue(hT, 0, (unsigned int)(size_t)&ldsH[0][0]);
#endif

    for (int s = 0; s < NSTAGE; ++s) {
        const int cur = s & 1;
        const int jt  = s * TJ;
#if USE_TDM
        __builtin_amdgcn_s_wait_tensorcnt(0);   // no-op on waves 1..3
        __syncthreads();                        // tile[cur] visible to all waves
        if (s + 1 < NSTAGE) {                   // prefetch tile s+1 into buffer cur^1
            const int jtn = jt + TJ;
            ldsF[cur ^ 1][tid] = f2[jtn + tid];
            if (wave == 0)
                tdm_issue(hT, jtn, (unsigned int)(size_t)&ldsH[cur ^ 1][0]);
        }
#else
        __syncthreads();
#pragma unroll
        for (int q = 0; q < 8; ++q) {
            int chunk = tid + q * 128;          // 0..1023, each = 8 halves
            int c  = chunk >> 4;                // feature column 0..63
            int j8 = (chunk & 15) << 3;         // j offset within tile
            *(uint4*)&ldsH[cur][c * TJP + j8] =
                *(const uint4*)&hT[c * N_ROWS + jt + j8];
        }
        ldsF[cur][tid] = f2[jt + tid];
        if (jt + TJ < N_ROWS)
            __builtin_prefetch(&hT[(tid >> 1) * N_ROWS + jt + TJ + (tid & 1) * 64], 0, 3);
        __syncthreads();
#endif

#pragma unroll
        for (int js = 0; js < TJ; js += 32) {
            // Build A (16x32 f16 weights) in-register in contiguous k order (same as B, not AMD's table order):
            // lane: M = lane%16 ; VGPR v: K = half*16 + 2v + {0,1}
            v16h A;
#pragma unroll
            for (int v = 0; v < 8; ++v) {
                int kb = (half << 4) + (v << 1);
                float e0 = f1i + ldsF[cur][js + kb];
                float e1 = f1i + ldsF[cur][js + kb + 1];
                e0 = (e0 > 0.f) ? e0 : GAT_ALPHA * e0;
                e1 = (e1 > 0.f) ? e1 : GAT_ALPHA * e1;
                float w0 = __expf(e0 - mi);     // <= 1.0, fp16-safe
                float w1 = __expf(e1 - mi);
                zacc += w0 + w1;
                A[2 * v]     = (_Float16)w0;
                A[2 * v + 1] = (_Float16)w1;
            }
            // 4 N-tiles of 16 output features each
#pragma unroll
            for (int t = 0; t < 4; ++t) {
                // B layout (32x16): lane N = lane%16, K = half*16 + 2v + {0,1}
                // -> 16 contiguous halves in LDS = two ds_load_b128
                const v8h* bp = (const v8h*)&ldsH[cur][(t * 16 + nlo) * TJP + js + half * 16];
                v16h B = __builtin_shufflevector(bp[0], bp[1],
                         0, 1, 2, 3, 4, 5, 6, 7, 8, 9, 10, 11, 12, 13, 14, 15);
                cacc[t] = __builtin_amdgcn_wmma_f32_16x16x32_f16(
                    false, A, false, B, (short)0, cacc[t], false, false);
            }
        }
    }

    // Z_i: lane l and lane l+16 each accumulated half the K values of row l%16.
    float zrow = zacc + __shfl_xor(zacc, 16, 32);  // lanes l, l+16 now hold Z of row l%16
    float Zr[8];
#pragma unroll
    for (int r = 0; r < 8; ++r)
        Zr[r] = __shfl(zrow, (half << 3) + r, 32);

    // C/D layout: lane holds N = lane%16, VGPR r -> M = r + 8*half
#pragma unroll
    for (int t = 0; t < 4; ++t) {
#pragma unroll
        for (int r = 0; r < 8; ++r) {
            int row = i0 + (half << 3) + r;
            float v = cacc[t][r] / Zr[r];
            v = (v > 0.f) ? v : (__expf(v) - 1.f);   // elu (alpha=1)
            out[row * F_OUT + t * 16 + nlo] = v;
        }
    }
}

extern "C" void kernel_launch(void* const* d_in, const int* in_sizes, int n_in,
                              void* d_out, int out_size, void* d_ws, size_t ws_size,
                              hipStream_t stream) {
    (void)in_sizes; (void)n_in; (void)out_size; (void)ws_size;
    const float* X = (const float*)d_in[0];   // 8192 x 128
    const float* W = (const float*)d_in[1];   // 128 x 64
    const float* a = (const float*)d_in[2];   // 128 x 1
    float* out = (float*)d_out;               // 8192 x 64

    char* ws = (char*)d_ws;
    float*    hf32 = (float*)ws;                                        // 2 MB
    _Float16* hT   = (_Float16*)(ws + (size_t)N_ROWS * F_OUT * 4);      // 1 MB
    float*    f1   = (float*)(ws + (size_t)3 * 1024 * 1024);            // 32 KB
    float*    f2   = f1 + N_ROWS;                                       // 32 KB
    float*    f2m  = f2 + N_ROWS;                                       // 4 B

    k_proj<<<(N_ROWS * F_OUT) / 256, 256, 0, stream>>>(X, W, hf32, hT);
    k_f12 <<<N_ROWS / 256, 256, 0, stream>>>(hf32, a, f1, f2);
    k_max <<<1, 256, 0, stream>>>(f2, f2m);
    k_attn<<<N_ROWS / 64, 128, 0, stream>>>(hT, f1, f2, f2m, out);
}
